// GAT_52862457479910
// MI455X (gfx1250) — matrix operand order repaired and checked by reading, not compiled, not run on hardware
//
#include <hip/hip_runtime.h>

// ---------------- problem constants ----------------
#define B_   16
#define N_   1024
#define F_   128
#define H_   8
#define D_   64
#define C_   16
#define FALPHA 0.2f

typedef __attribute__((ext_vector_type(16))) __bf16 v16bf;
typedef __attribute__((ext_vector_type(8)))  float  v8f;

static __device__ inline v8f zero8() {
  v8f z;
#pragma unroll
  for (int i = 0; i < 8; ++i) z[i] = 0.0f;
  return z;
}

static __device__ inline v8f wmma_bf16(v16bf a, v16bf b, v8f c) {
  // D = A(16x32 bf16) * B(32x16 bf16) + C(16x16 f32)
  return __builtin_amdgcn_wmma_f32_16x16x32_bf16(
      /*neg_a=*/false, a, /*neg_b=*/false, b,
      /*c_mod=*/(short)0, c, /*reuse_a=*/false, /*reuse_b=*/false);
}

// A-tile (16x32) from a row-major fp32 row pointer; lane supplies two runs of
// 8 consecutive columns: K = 8h+e (e<8) and K = 16+8h+(e-8).
static __device__ inline v16bf load_a_f32(const float* row, int k0, int h) {
  const float4* p0 = (const float4*)(row + k0 + 8 * h);
  const float4* p1 = (const float4*)(row + k0 + 16 + 8 * h);
  float4 a0 = p0[0], a1 = p0[1], b0 = p1[0], b1 = p1[1];
  v16bf r;
  r[0] = (__bf16)a0.x; r[1] = (__bf16)a0.y; r[2] = (__bf16)a0.z; r[3] = (__bf16)a0.w;
  r[4] = (__bf16)a1.x; r[5] = (__bf16)a1.y; r[6] = (__bf16)a1.z; r[7] = (__bf16)a1.w;
  r[8]  = (__bf16)b0.x; r[9]  = (__bf16)b0.y; r[10] = (__bf16)b0.z; r[11] = (__bf16)b0.w;
  r[12] = (__bf16)b1.x; r[13] = (__bf16)b1.y; r[14] = (__bf16)b1.z; r[15] = (__bf16)b1.w;
  return r;
}

// A-tile from a row-major bf16 row pointer (two 16B loads).
static __device__ inline v16bf load_a_bf16(const __bf16* row, int k0, int h) {
  v16bf r;
  ((uint4*)&r)[0] = *(const uint4*)(row + k0 + 8 * h);
  ((uint4*)&r)[1] = *(const uint4*)(row + k0 + 16 + 8 * h);
  return r;
}

// B-tile (32x16) from a K-transposed bf16 matrix bt[n][k] with leading dim ldb:
// lane (n = l%16, h = l/16) needs K = k0+8h+e (e<8) and k0+16+8h+(e-8).
static __device__ inline v16bf load_b_bt(const __bf16* bt, int ldb, int d0,
                                         int k0, int n, int h) {
  const __bf16* p = bt + (size_t)(d0 + n) * ldb + k0 + 8 * h;
  v16bf r;
  ((uint4*)&r)[0] = *(const uint4*)(p);
  ((uint4*)&r)[1] = *(const uint4*)(p + 16);
  return r;
}

// ---------------- kernel 1: pack adjacency to bitmask ----------------
__global__ __launch_bounds__(256) void pack_mask(const int* __restrict__ adj,
                                                 unsigned* __restrict__ maskp) {
  int idx = blockIdx.x * 256 + threadIdx.x;           // word index
  if (idx >= B_ * N_ * (N_ / 32)) return;
  int w = idx & 31;
  int i = (idx >> 5) & (N_ - 1);
  int b = idx >> (5 + 10);
  const int* p = adj + ((size_t)b * N_ + i) * N_ + w * 32;
  unsigned m = 0;
#pragma unroll
  for (int k = 0; k < 32; ++k) m |= (p[k] > 0 ? 1u : 0u) << k;
  maskp[idx] = m;
}

// ---------------- kernel 2: transpose weights to bf16 ----------------
__global__ __launch_bounds__(256) void prep_weights(const float* __restrict__ W,
                                                    const float* __restrict__ Wo,
                                                    __bf16* __restrict__ WT,
                                                    __bf16* __restrict__ WoT) {
  int idx = blockIdx.x * 256 + threadIdx.x;
  const int nWT = H_ * D_ * F_;                       // 65536
  if (idx < nWT) {
    int k = idx % F_;
    int d = (idx / F_) % D_;
    int h = idx / (F_ * D_);
    WT[idx] = (__bf16)W[((size_t)h * F_ + k) * D_ + d];
  } else {
    int idx2 = idx - nWT;
    if (idx2 < C_ * (H_ * D_)) {
      int k = idx2 % (H_ * D_);
      int c = idx2 / (H_ * D_);
      WoT[idx2] = (__bf16)Wo[(size_t)k * C_ + c];
    }
  }
}

// ---------------- kernel 3: Wh = x@W  (+f1,f2, WhT bf16) ----------------
__global__ __launch_bounds__(32) void whl1(const float* __restrict__ xs,
                                           const __bf16* __restrict__ WT,
                                           const float* __restrict__ a1,
                                           const float* __restrict__ a2,
                                           __bf16* __restrict__ WhT,
                                           float* __restrict__ f1,
                                           float* __restrict__ f2) {
  int blk = blockIdx.x;                                // (b, h, rowtile)
  int rt = blk & 63;
  int h  = (blk >> 6) & (H_ - 1);
  int b  = blk >> 9;
  int n0 = rt * 16;
  int l = threadIdx.x, lh = l >> 4, lm = l & 15;

  const float*  xrow = xs + ((size_t)b * N_ + n0 + lm) * F_;
  const __bf16* wt   = WT + (size_t)h * D_ * F_;

  v8f acc[4];
#pragma unroll
  for (int t = 0; t < 4; ++t) acc[t] = zero8();

#pragma unroll
  for (int k0 = 0; k0 < F_; k0 += 32) {
    v16bf A = load_a_f32(xrow, k0, lh);
#pragma unroll
    for (int t = 0; t < 4; ++t) {
      v16bf Bm = load_b_bt(wt, F_, 16 * t, k0, lm, lh);
      acc[t] = wmma_bf16(A, Bm, acc[t]);
    }
  }

  __shared__ float tile[16 * 65];
#pragma unroll
  for (int t = 0; t < 4; ++t)
#pragma unroll
    for (int v = 0; v < 8; ++v) {
      int M = v + 8 * lh;
      tile[M * 65 + 16 * t + lm] = acc[t][v];
    }
  __syncthreads();

  if (l < 16) {
    const float* av1 = a1 + h * D_;
    const float* av2 = a2 + h * D_;
    float s1 = 0.f, s2 = 0.f;
    for (int d = 0; d < D_; ++d) {
      float w = tile[l * 65 + d];
      s1 += w * av1[d];
      s2 += w * av2[d];
    }
    size_t o = ((size_t)b * H_ + h) * N_ + n0 + l;
    f1[o] = s1;
    f2[o] = s2;
  }

  __bf16* wh = WhT + (size_t)(b * H_ + h) * D_ * N_;
  for (int dd = l; dd < D_; dd += 32) {
    v16bf vb;
#pragma unroll
    for (int M = 0; M < 16; ++M) vb[M] = (__bf16)tile[M * 65 + dd];
    uint4* dst = (uint4*)(wh + (size_t)dd * N_ + n0);
    dst[0] = ((uint4*)&vb)[0];
    dst[1] = ((uint4*)&vb)[1];
  }
}

// ---------------- kernel 4: fused attention layer 1 -> x1 bf16 ----------------
__global__ __launch_bounds__(128) void attn1(const float* __restrict__ f1g,
                                             const float* __restrict__ f2g,
                                             const unsigned* __restrict__ maskp,
                                             const __bf16* __restrict__ WhT,
                                             __bf16* __restrict__ x1) {
  int blk = blockIdx.x;                                // (b, h, rb) ; 64 rows per block
  int rb = blk & 15;
  int h  = (blk >> 4) & (H_ - 1);
  int b  = blk >> 7;
  int tid = threadIdx.x, wave = tid >> 5, l = tid & 31, lh = l >> 4, lm = l & 15;
  int n0 = rb * 64 + wave * 16;

  __shared__ float f2s[N_];
  __shared__ float rs[4][32];

  const float* f2p = f2g + ((size_t)b * H_ + h) * N_;
  for (int i = tid; i < N_; i += 128) f2s[i] = f2p[i];
  __syncthreads();

  float fi = f1g[((size_t)b * H_ + h) * N_ + n0 + lm];
  const unsigned* mrow = maskp + ((size_t)b * N_ + n0 + lm) * (N_ / 32);
  const __bf16* wh = WhT + (size_t)(b * H_ + h) * D_ * N_;

  v8f acc[4];
#pragma unroll
  for (int t = 0; t < 4; ++t) acc[t] = zero8();
  float rsum = 0.f;

  for (int j0 = 0; j0 < N_; j0 += 32) {
    unsigned mw = mrow[j0 >> 5];
    float4 q0 = *(const float4*)&f2s[j0 + 8 * lh];
    float4 q1 = *(const float4*)&f2s[j0 + 8 * lh + 4];
    float4 q2 = *(const float4*)&f2s[j0 + 16 + 8 * lh];
    float4 q3 = *(const float4*)&f2s[j0 + 16 + 8 * lh + 4];
    float f2v[16] = {q0.x, q0.y, q0.z, q0.w, q1.x, q1.y, q1.z, q1.w,
                     q2.x, q2.y, q2.z, q2.w, q3.x, q3.y, q3.z, q3.w};
    v16bf P;
#pragma unroll
    for (int e = 0; e < 16; ++e) {
      int K = (e < 8) ? (8 * lh + e) : (16 + 8 * lh + (e - 8));
      float s = fi + f2v[e];
      s = s > 0.f ? s : FALPHA * s;
      float p = ((mw >> K) & 1u) ? __expf(s) : 0.f;
      rsum += p;
      P[e] = (__bf16)p;
    }
#pragma unroll
    for (int t = 0; t < 4; ++t) {
      v16bf Bm = load_b_bt(wh, N_, 16 * t, j0, lm, lh);
      acc[t] = wmma_bf16(P, Bm, acc[t]);
    }
  }

  rs[wave][l] = rsum;
  __syncthreads();

  __bf16* xo = x1 + (size_t)b * N_ * (H_ * D_);
#pragma unroll
  for (int t = 0; t < 4; ++t)
#pragma unroll
    for (int v = 0; v < 8; ++v) {
      int M = v + 8 * lh;
      float lsum = rs[wave][M] + rs[wave][M + 16];
      float hv = acc[t][v] / lsum;
      float ev = hv > 0.f ? hv : (__expf(hv) - 1.f);   // elu
      xo[(size_t)(n0 + M) * (H_ * D_) + h * D_ + 16 * t + lm] = (__bf16)ev;
    }
}

// ---------------- kernel 5: Wh2 = x1@Wo (+g1,g2, Wh2T bf16) ----------------
__global__ __launch_bounds__(32) void wh2k(const __bf16* __restrict__ x1,
                                           const __bf16* __restrict__ WoT,
                                           const float* __restrict__ ao1,
                                           const float* __restrict__ ao2,
                                           __bf16* __restrict__ Wh2T,
                                           float* __restrict__ g1,
                                           float* __restrict__ g2) {
  int blk = blockIdx.x;
  int rt = blk & 63;
  int b  = blk >> 6;
  int n0 = rt * 16;
  int l = threadIdx.x, lh = l >> 4, lm = l & 15;

  const __bf16* xrow = x1 + ((size_t)b * N_ + n0 + lm) * (H_ * D_);
  v8f acc = zero8();
#pragma unroll 4
  for (int k0 = 0; k0 < H_ * D_; k0 += 32) {
    v16bf A  = load_a_bf16(xrow, k0, lh);
    v16bf Bm = load_b_bt(WoT, H_ * D_, 0, k0, lm, lh);
    acc = wmma_bf16(A, Bm, acc);
  }

  __shared__ float tile[16 * 17];
#pragma unroll
  for (int v = 0; v < 8; ++v) {
    int M = v + 8 * lh;
    tile[M * 17 + lm] = acc[v];
  }
  __syncthreads();

  if (l < 16) {
    float s1 = 0.f, s2 = 0.f;
    for (int c = 0; c < C_; ++c) {
      float w = tile[l * 17 + c];
      s1 += w * ao1[c];
      s2 += w * ao2[c];
    }
    size_t o = (size_t)b * N_ + n0 + l;
    g1[o] = s1;
    g2[o] = s2;
    v16bf vb;
#pragma unroll
    for (int M = 0; M < 16; ++M) vb[M] = (__bf16)tile[M * 17 + l];
    uint4* dst = (uint4*)(Wh2T + ((size_t)b * C_ + l) * N_ + n0);
    dst[0] = ((uint4*)&vb)[0];
    dst[1] = ((uint4*)&vb)[1];
  }
}

// ---------------- kernel 6: fused attention layer 2 -> h2 fp32 ----------------
__global__ __launch_bounds__(128) void attn2(const float* __restrict__ g1,
                                             const float* __restrict__ g2,
                                             const unsigned* __restrict__ maskp,
                                             const __bf16* __restrict__ Wh2T,
                                             float* __restrict__ h2) {
  int blk = blockIdx.x;                                // (b, rb)
  int rb = blk & 15;
  int b  = blk >> 4;
  int tid = threadIdx.x, wave = tid >> 5, l = tid & 31, lh = l >> 4, lm = l & 15;
  int n0 = rb * 64 + wave * 16;

  __shared__ float g2s[N_];
  __shared__ float rs[4][32];
  for (int i = tid; i < N_; i += 128) g2s[i] = g2[(size_t)b * N_ + i];
  __syncthreads();

  float gi = g1[(size_t)b * N_ + n0 + lm];
  const unsigned* mrow = maskp + ((size_t)b * N_ + n0 + lm) * (N_ / 32);
  const __bf16* wt = Wh2T + (size_t)b * C_ * N_;

  v8f acc = zero8();
  float rsum = 0.f;
  for (int j0 = 0; j0 < N_; j0 += 32) {
    unsigned mw = mrow[j0 >> 5];
    float4 q0 = *(const float4*)&g2s[j0 + 8 * lh];
    float4 q1 = *(const float4*)&g2s[j0 + 8 * lh + 4];
    float4 q2 = *(const float4*)&g2s[j0 + 16 + 8 * lh];
    float4 q3 = *(const float4*)&g2s[j0 + 16 + 8 * lh + 4];
    float gv[16] = {q0.x, q0.y, q0.z, q0.w, q1.x, q1.y, q1.z, q1.w,
                    q2.x, q2.y, q2.z, q2.w, q3.x, q3.y, q3.z, q3.w};
    v16bf P;
#pragma unroll
    for (int e = 0; e < 16; ++e) {
      int K = (e < 8) ? (8 * lh + e) : (16 + 8 * lh + (e - 8));
      float s = gi + gv[e];
      s = s > 0.f ? s : FALPHA * s;
      float p = ((mw >> K) & 1u) ? __expf(s) : 0.f;
      rsum += p;
      P[e] = (__bf16)p;
    }
    v16bf Bm = load_b_bt(wt, N_, 0, j0, lm, lh);
    acc = wmma_bf16(P, Bm, acc);
  }

  rs[wave][l] = rsum;
  __syncthreads();

#pragma unroll
  for (int v = 0; v < 8; ++v) {
    int M = v + 8 * lh;
    float lsum = rs[wave][M] + rs[wave][M + 16];
    h2[((size_t)b * N_ + n0 + M) * C_ + lm] = acc[v] / lsum;
  }
}

// ---------------- kernel 7: pred = mean_n( elu(h2) @ Wp ) + bp ----------------
__global__ __launch_bounds__(256) void finalk(const float* __restrict__ h2,
                                              const float* __restrict__ Wp,
                                              const float* __restrict__ bp,
                                              float* __restrict__ out) {
  int b = blockIdx.x;
  int tid = threadIdx.x;
  int c = tid & 15;
  int r = tid >> 4;                                    // 16 row groups
  __shared__ float wps[C_ * C_];
  if (tid < C_ * C_) wps[tid] = Wp[tid];
  __syncthreads();

  float acc = 0.f;
  for (int n = r; n < N_; n += 16) {
    const float* hrow = h2 + ((size_t)b * N_ + n) * C_;
    float s = 0.f;
#pragma unroll
    for (int k = 0; k < C_; ++k) {
      float x = hrow[k];
      x = x > 0.f ? x : (__expf(x) - 1.f);             // outer elu
      s += x * wps[k * C_ + c];
    }
    acc += s;
  }

  __shared__ float red[256];
  red[tid] = acc;
  __syncthreads();
  for (int s = 128; s >= 16; s >>= 1) {
    if (tid < s) red[tid] += red[tid + s];
    __syncthreads();
  }
  if (tid < 16) out[b * C_ + tid] = red[tid] / (float)N_ + bp[tid];
}

// ---------------- launcher ----------------
extern "C" void kernel_launch(void* const* d_in, const int* in_sizes, int n_in,
                              void* d_out, int out_size, void* d_ws, size_t ws_size,
                              hipStream_t stream) {
  const float* xs  = (const float*)d_in[0];
  const int*   adj = (const int*)d_in[1];
  const float* W   = (const float*)d_in[2];
  const float* a1  = (const float*)d_in[3];
  const float* a2  = (const float*)d_in[4];
  const float* Wo  = (const float*)d_in[5];
  const float* ao1 = (const float*)d_in[6];
  const float* ao2 = (const float*)d_in[7];
  const float* Wp  = (const float*)d_in[8];
  const float* bp  = (const float*)d_in[9];
  float* out = (float*)d_out;

  char* ws = (char*)d_ws;
  size_t off = 0;
  auto take = [&](size_t bytes) {
    char* p = ws + off;
    off = (off + bytes + 255) & ~(size_t)255;
    return p;
  };
  __bf16*   WhT   = (__bf16*)take((size_t)B_ * H_ * D_ * N_ * 2);   // 16 MB
  float*    f1    = (float*)take((size_t)B_ * H_ * N_ * 4);
  float*    f2    = (float*)take((size_t)B_ * H_ * N_ * 4);
  unsigned* maskp = (unsigned*)take((size_t)B_ * N_ * (N_ / 32) * 4); // 2 MB
  __bf16*   x1    = (__bf16*)take((size_t)B_ * N_ * H_ * D_ * 2);   // 16 MB
  __bf16*   Wh2T  = (__bf16*)take((size_t)B_ * C_ * N_ * 2);
  float*    g1    = (float*)take((size_t)B_ * N_ * 4);
  float*    g2    = (float*)take((size_t)B_ * N_ * 4);
  float*    h2    = (float*)take((size_t)B_ * N_ * C_ * 4);         // 1 MB
  __bf16*   WT    = (__bf16*)take((size_t)H_ * D_ * F_ * 2);
  __bf16*   WoT   = (__bf16*)take((size_t)C_ * H_ * D_ * 2);
  (void)ws_size; (void)in_sizes; (void)n_in; (void)out_size;

  {
    int words = B_ * N_ * (N_ / 32);
    pack_mask<<<(words + 255) / 256, 256, 0, stream>>>(adj, maskp);
  }
  {
    int tot = H_ * D_ * F_ + C_ * H_ * D_;
    prep_weights<<<(tot + 255) / 256, 256, 0, stream>>>(W, Wo, WT, WoT);
  }
  whl1<<<B_ * H_ * (N_ / 16), 32, 0, stream>>>(xs, WT, a1, a2, WhT, f1, f2);
  attn1<<<B_ * H_ * 16, 128, 0, stream>>>(f1, f2, maskp, WhT, x1);
  wh2k<<<B_ * (N_ / 16), 32, 0, stream>>>(x1, WoT, ao1, ao2, Wh2T, g1, g2);
  attn2<<<B_ * 16, 128, 0, stream>>>(g1, g2, maskp, Wh2T, h2);
  finalk<<<B_, 256, 0, stream>>>(h2, Wp, bp, out);
}
